// BaseModel_83176336654779
// MI455X (gfx1250) — matrix operand order repaired and checked by reading, not compiled, not run on hardware
//
#include <hip/hip_runtime.h>

typedef _Float16 v8h  __attribute__((ext_vector_type(8)));
typedef _Float16 v16h __attribute__((ext_vector_type(16)));
typedef float    v8f  __attribute__((ext_vector_type(8)));

#define B_  256
#define T_  4096
#define F_  32
#define H_  128
#define G_  512   // 4*H

__device__ __forceinline__ float fast_sigmoid(float x) {
    return 1.0f / (1.0f + __expf(-x));
}
__device__ __forceinline__ float fast_tanh(float x) {
    // tanh(x) = 2*sigmoid(2x) - 1
    return 2.0f / (1.0f + __expf(-2.0f * x)) - 1.0f;
}

__launch_bounds__(512, 1)
__global__ void lstm_wmma_kernel(const float* __restrict__ inp,      // (B,T,F) f32
                                 const int*   __restrict__ seq_lens, // (B,)
                                 const float* __restrict__ W,        // (F, 4H) f32
                                 const float* __restrict__ U,        // (H, 4H) f32
                                 const float* __restrict__ bias,     // (4H,)
                                 const float* __restrict__ Wf,       // (H,1)
                                 const float* __restrict__ bf,       // (1,)
                                 float*       __restrict__ out)      // (B,)
{
    // LDS: x tile (16x32 f16), h tile (16x128 f16), z buffer (16x512 f32)
    __shared__ alignas(16) _Float16 hX[16 * F_];   // 1 KB
    __shared__ alignas(16) _Float16 hH[16 * H_];   // 4 KB
    __shared__ alignas(16) float    zb[16 * G_];   // 32 KB

    const int tid  = threadIdx.x;
    const int lane = tid & 31;
    const int wv   = tid >> 5;       // wave id 0..15, owns gate cols [wv*32, wv*32+32)
    const int g    = lane >> 4;      // half-wave group (0/1)
    const int ml   = lane & 15;      // row (A) or col (B, C/D) index within 16
    const int b0   = blockIdx.x * 16;

    // ---------------- one-time: stage W, U slices into VGPRs (f32 -> f16) -------------
    // B-operand layout (32x16, f16): lane = g*16 + N, halves e: K = kbase + g*16 + e.
    v16h  Bx[2];        // W  B-operands, one per owned N-tile (K = 0..31)
    v16h  BU[4][2];     // U  B-operands: [ktile 0..3][N-tile]
    float biasv[2];
    for (int p = 0; p < 2; ++p) {
        const int ncol = (wv * 2 + p) * 16 + ml;
        biasv[p] = bias[ncol];
        #pragma unroll
        for (int e = 0; e < 16; ++e)
            Bx[p][e] = (_Float16)W[(g * 16 + e) * G_ + ncol];
        #pragma unroll
        for (int kt = 0; kt < 4; ++kt)
            #pragma unroll
            for (int e = 0; e < 16; ++e)
                BU[kt][p][e] = (_Float16)U[(kt * 32 + g * 16 + e) * G_ + ncol];
    }

    // ---------------- init h = 0 in LDS ----------------
    for (int i = tid; i < 16 * H_; i += 512) hH[i] = (_Float16)0.0f;

    // elementwise ownership: thread -> (batch row be, 4 hidden cols j = lane + 32q)
    // strided by 32 so consecutive lanes hit consecutive LDS words (no bank conflicts)
    const int be = tid >> 5;          // 0..15 (one batch row per wave)
    const int jl = tid & 31;          // base hidden col within row
    const int sl = seq_lens[b0 + be];
    float c[4] = {0.0f, 0.0f, 0.0f, 0.0f};

    // x loader ownership: thread -> (batch row be, feature fx)
    const int fx = tid & 31;
    const float* xptr = inp + (size_t)(b0 + be) * T_ * F_ + fx;

    // software pipeline: xreg holds x_t at the top of step t; x_{t+1} is loaded
    // AFTER the first barrier (inside the GEMM region) so that (a) its WAR hazard
    // against the ds_store of x_t is already covered by the barrier's dscnt wait,
    // and (b) its latency is hidden behind WMMA + gate math of step t.
    float xreg = xptr[0];

    for (int t = 0; t < T_; ++t) {
        // ---- stage x_t tile into LDS (resident in xreg since last step) ----
        hX[be * F_ + fx] = (_Float16)xreg;
        const int tp = (t + 2 < T_) ? (t + 2) : (T_ - 1);
        __builtin_prefetch(xptr + (size_t)tp * F_, 0, 1);
        __syncthreads();   // hH (prev step) + hX ready; dscnt drained here

        // issue x_{t+1} load now -- retires during WMMA + gates below
        const int tn = (t + 1 < T_) ? (t + 1) : (T_ - 1);
        const float xnext = xptr[(size_t)tn * F_];

        // ---- A operands from LDS (layout: lane g*16+M, halves: K kb+e / kb+8+e) ----
        const int kb = g * 16;
        v8h xlo = *(const v8h*)&hX[ml * F_ + kb];
        v8h xhi = *(const v8h*)&hX[ml * F_ + kb + 8];
        v16h Ax = __builtin_shufflevector(xlo, xhi,
                    0,1,2,3,4,5,6,7,8,9,10,11,12,13,14,15);
        v16h Ah[4];
        #pragma unroll
        for (int kt = 0; kt < 4; ++kt) {
            v8h lo = *(const v8h*)&hH[ml * H_ + kt * 32 + kb];
            v8h hi = *(const v8h*)&hH[ml * H_ + kt * 32 + kb + 8];
            Ah[kt] = __builtin_shufflevector(lo, hi,
                        0,1,2,3,4,5,6,7,8,9,10,11,12,13,14,15);
        }

        // ---- z = x@W + h@U + b for this wave's two 16-col tiles ----
        #pragma unroll
        for (int p = 0; p < 2; ++p) {
            v8f acc;
            #pragma unroll
            for (int r = 0; r < 8; ++r) acc[r] = biasv[p];   // bias preload into C
            acc = __builtin_amdgcn_wmma_f32_16x16x32_f16(
                      false, Ax, false, Bx[p], (short)0, acc, false, false);
            #pragma unroll
            for (int kt = 0; kt < 4; ++kt)
                acc = __builtin_amdgcn_wmma_f32_16x16x32_f16(
                          false, Ah[kt], false, BU[kt][p], (short)0, acc, false, false);
            // C/D layout: VGPR r -> row r + g*8, lane&15 -> col
            const int ncol = (wv * 2 + p) * 16 + ml;
            #pragma unroll
            for (int r = 0; r < 8; ++r)
                zb[(r + g * 8) * G_ + ncol] = acc[r];
        }

        // Pin the schedule of the GEMM region so the scheduler does not
        // serialize LDS-load latency with WMMA pairs:
        //   x-load first (whole region to retire), then ALL A-operand DS reads
        //   batched (single LDS latency, staggered dscnt waits), then the 10
        //   WMMAs, then the 8 packed z stores.
        __builtin_amdgcn_sched_group_barrier(0x020, 1,  0);  // VMEM read (x_{t+1})
        __builtin_amdgcn_sched_group_barrier(0x100, 10, 0);  // DS reads (A operands)
        __builtin_amdgcn_sched_group_barrier(0x008, 10, 0);  // WMMA
        __builtin_amdgcn_sched_group_barrier(0x200, 8,  0);  // DS writes (z)

        __syncthreads();   // z complete

        // ---- gates + state update (mask: only commit if t < seq_len) ----
        if (t < sl) {
            #pragma unroll
            for (int q = 0; q < 4; ++q) {
                const int j  = jl + 32 * q;            // conflict-free stride
                const float zi = zb[be * G_ + j];
                const float zf = zb[be * G_ + j + H_];
                const float zg = zb[be * G_ + j + 2 * H_];
                const float zo = zb[be * G_ + j + 3 * H_];
                const float ig = fast_sigmoid(zi);
                const float fg = fast_sigmoid(zf);
                const float gg = fast_tanh(zg);
                const float og = fast_sigmoid(zo);
                const float cn = fg * c[q] + ig * gg;
                c[q] = cn;
                hH[be * H_ + j] = (_Float16)(og * fast_tanh(cn));
            }
        }
        xreg = xnext;
    }
    __syncthreads();

    // ---- final projection: out[b] = h[b,:] . Wf + bf ----
    if (tid < 16) {
        float s = bf[0];
        #pragma unroll 8
        for (int j = 0; j < H_; ++j)
            s += (float)hH[tid * H_ + j] * Wf[j];
        out[b0 + tid] = s;
    }
}

extern "C" void kernel_launch(void* const* d_in, const int* in_sizes, int n_in,
                              void* d_out, int out_size, void* d_ws, size_t ws_size,
                              hipStream_t stream) {
    const float* inp = (const float*)d_in[0];
    const int*   sl  = (const int*)  d_in[1];
    const float* W   = (const float*)d_in[2];
    const float* U   = (const float*)d_in[3];
    const float* b   = (const float*)d_in[4];
    const float* Wf  = (const float*)d_in[5];
    const float* bf  = (const float*)d_in[6];
    float*       out = (float*)d_out;
    (void)in_sizes; (void)n_in; (void)out_size; (void)d_ws; (void)ws_size;

    lstm_wmma_kernel<<<dim3(B_ / 16), dim3(512), 0, stream>>>(
        inp, sl, W, U, b, Wf, bf, out);
}
